// GraphCut_74234214744332
// MI455X (gfx1250) — matrix operand order repaired and checked by reading, not compiled, not run on hardware
//
#include <hip/hip_runtime.h>
#include <hip/hip_bf16.h>
#include <stdint.h>

// Reference: M=8192, K=128, TEMPERATURE=0.2, LAMDA=1.0, NUM_CLASSES=21
#define M_ROWS   8192
#define K_DIM    128
#define NTILE    (M_ROWS / 16)   // 512 row/col tiles
#define TEMP_INV 5.0f            // 1 / 0.2
#define LAMDA_C  1.0f

typedef __attribute__((ext_vector_type(16))) __bf16 v16bf;
typedef __attribute__((ext_vector_type(8)))  float  v8f;

// register-level container: one WMMA bf16 fragment = 8 VGPRs = 2 x uint4
union Frag { v16bf v; uint4 q[2]; };

__device__ __forceinline__ unsigned short f32_to_bf16_rne(float x) {
    union { float f; uint32_t u; } c; c.f = x;
    uint32_t u = c.u;
    uint32_t r = u + 0x7FFFu + ((u >> 16) & 1u);   // round-to-nearest-even
    return (unsigned short)(r >> 16);
}
__device__ __forceinline__ float bf16_to_f32(unsigned short h) {
    union { uint32_t u; float f; } c; c.u = ((uint32_t)h) << 16;
    return c.f;
}

// ---------------------------------------------------------------------------
// Kernel 1: split fp32 features into bf16 hi + bf16 lo (bf16x3 emulation)
// ---------------------------------------------------------------------------
__global__ void split_bf16_kernel(const float* __restrict__ f,
                                  unsigned short* __restrict__ hi,
                                  unsigned short* __restrict__ lo, int n) {
    int i = blockIdx.x * blockDim.x + threadIdx.x;
    if (i < n) {
        float x = f[i];
        unsigned short h = f32_to_bf16_rne(x);
        hi[i] = h;
        lo[i] = f32_to_bf16_rne(x - bf16_to_f32(h));
    }
}

// ---------------------------------------------------------------------------
// Kernel 2: fused  S = F F^T / T  +  online row-max  +  masked exp sums.
// One block = one 16-row tile; 8 waves sweep the 512 column tiles.
//
// WMMA bf16 16x16x32 fragment layouts (wave32):
//   A (16x32, M x K): lane L<16 -> row M=L, K {8h..8h+7} in v0..3 and
//                     K {16+8h..16+8h+7} in v4..7 where h = L>=16 (khalf).
//   B (32x16, K x N): lane L -> col N=L%16, same K order as A: K {8h..8h+7} in v0..3,
//                     K {16+8h..16+8h+7} in v4..7, read from F's row for that column.
//   C/D (16x16 f32):  acc[d] = S[d + 8*khalf][L%16].
// ---------------------------------------------------------------------------
__global__ __launch_bounds__(256, 1)
void supcon_loss_kernel(const unsigned short* __restrict__ hi,
                        const unsigned short* __restrict__ lo,
                        const int* __restrict__ labels,
                        float* __restrict__ partials) {
    __shared__ float red_m[8][16];
    __shared__ float red_p[8][16];
    __shared__ float red_n[8][16];
    __shared__ float rowv[16];

    const int ri    = blockIdx.x;          // row tile index
    const int tid   = threadIdx.x;
    const int lane  = tid & 31;
    const int wave  = tid >> 5;            // 0..7
    const int khalf = lane >> 4;           // 0/1 : which K-half this lane holds
    const int lcol  = lane & 15;

    // ---- A fragments for this row tile (hi and lo), K = 4 chunks of 32 ----
    const int rowA = ri * 16 + lcol;       // A-fragment row index = lane%16
    Frag Ahi[4], Alo[4];
#pragma unroll
    for (int c = 0; c < 4; ++c) {
        const unsigned short* ph = hi + rowA * K_DIM + 32 * c + khalf * 8;
        const unsigned short* pl = lo + rowA * K_DIM + 32 * c + khalf * 8;
        Ahi[c].q[0] = *(const uint4*)(ph);        // K = 32c+8h+0..7
        Ahi[c].q[1] = *(const uint4*)(ph + 16);   // K = 32c+16+8h+0..7
        Alo[c].q[0] = *(const uint4*)(pl);
        Alo[c].q[1] = *(const uint4*)(pl + 16);
    }

    // labels of the 8 rows this lane accumulates (rows 8*khalf + d)
    int rowlab[8];
#pragma unroll
    for (int d = 0; d < 8; ++d) rowlab[d] = labels[ri * 16 + 8 * khalf + d];

    // online softmax-style running state per row
    float mrun[8], sp[8], sn[8];
#pragma unroll
    for (int d = 0; d < 8; ++d) { mrun[d] = -__builtin_inff(); sp[d] = 0.f; sn[d] = 0.f; }

    for (int cj = wave; cj < NTILE; cj += 8) {
        const int colB = cj * 16 + lcol;   // this lane's B column (= gcol for acc)
        const int clab = labels[colB];

        Frag Bhi[4], Blo[4];
#pragma unroll
        for (int c = 0; c < 4; ++c) {
            const unsigned short* ph = hi + colB * K_DIM + 32 * c + khalf * 8;
            const unsigned short* pl = lo + colB * K_DIM + 32 * c + khalf * 8;
            Bhi[c].q[0] = *(const uint4*)(ph);      // K = 32c+8h+0..7
            Bhi[c].q[1] = *(const uint4*)(ph + 16); // K = 32c+16+8h+0..7
            Blo[c].q[0] = *(const uint4*)(pl);
            Blo[c].q[1] = *(const uint4*)(pl + 16);
        }
        if (cj + 8 < NTILE) {  // prefetch next tile's B rows (global_prefetch_b8)
            __builtin_prefetch(hi + (cj + 8) * 16 * K_DIM + lcol * K_DIM, 0, 1);
            __builtin_prefetch(lo + (cj + 8) * 16 * K_DIM + lcol * K_DIM, 0, 1);
        }

        // S_tile = hi*hi + hi*lo + lo*hi  (bf16x3, ~fp32 accuracy), f32 accum
        v8f acc = {};
#pragma unroll
        for (int c = 0; c < 4; ++c) {
            acc = __builtin_amdgcn_wmma_f32_16x16x32_bf16(false, Ahi[c].v, false, Bhi[c].v,
                                                          (short)0, acc, false, false);
            acc = __builtin_amdgcn_wmma_f32_16x16x32_bf16(false, Ahi[c].v, false, Blo[c].v,
                                                          (short)0, acc, false, false);
            acc = __builtin_amdgcn_wmma_f32_16x16x32_bf16(false, Alo[c].v, false, Bhi[c].v,
                                                          (short)0, acc, false, false);
        }

        // online update: max includes diagonal (reference maxes full row),
        // sums exclude it.
#pragma unroll
        for (int d = 0; d < 8; ++d) {
            const float sim = acc[d] * TEMP_INV;
            const int   gr  = ri * 16 + 8 * khalf + d;
            const float nm  = fmaxf(mrun[d], sim);
            const float sc  = __expf(mrun[d] - nm);   // exp(-inf)=0 on first hit
            const float e   = __expf(sim - nm);
            const bool  diag = (gr == colB);
            const bool  same = (rowlab[d] == clab);
            sp[d] = sp[d] * sc + ((!diag) && same  ? e : 0.f);
            sn[d] = sn[d] * sc + ((!diag) && !same ? e : 0.f);
            mrun[d] = nm;
        }
    }

    // ---- butterfly reduce across the 16 lanes of each half-wave ----
#pragma unroll
    for (int off = 8; off >= 1; off >>= 1) {
#pragma unroll
        for (int d = 0; d < 8; ++d) {
            float om = __shfl_xor(mrun[d], off, 32);
            float op = __shfl_xor(sp[d],   off, 32);
            float on = __shfl_xor(sn[d],   off, 32);
            float nm = fmaxf(mrun[d], om);
            float s0 = __expf(mrun[d] - nm), s1 = __expf(om - nm);
            sp[d] = sp[d] * s0 + op * s1;
            sn[d] = sn[d] * s0 + on * s1;
            mrun[d] = nm;
        }
    }

    if (lcol == 0) {
#pragma unroll
        for (int d = 0; d < 8; ++d) {
            red_m[wave][8 * khalf + d] = mrun[d];
            red_p[wave][8 * khalf + d] = sp[d];
            red_n[wave][8 * khalf + d] = sn[d];
        }
    }
    __syncthreads();

    // combine the 8 waves' partial triples in fixed order (deterministic)
    if (tid < 16) {
        float m = red_m[0][tid], p = red_p[0][tid], n = red_n[0][tid];
        for (int w = 1; w < 8; ++w) {
            float om = red_m[w][tid], op = red_p[w][tid], on = red_n[w][tid];
            float nm = fmaxf(m, om);
            float s0 = __expf(m - nm), s1 = __expf(om - nm);
            p = p * s0 + op * s1;
            n = n * s0 + on * s1;
            m = nm;
        }
        rowv[tid] = LAMDA_C * p - n;       // log_prob for this row
    }
    __syncthreads();
    if (tid == 0) {
        float s = 0.f;
        for (int r = 0; r < 16; ++r) s += rowv[r];
        partials[ri] = s;
    }
}

// ---------------------------------------------------------------------------
// Kernel 3: deterministic fixed-order sum of 512 block partials -> loss
// ---------------------------------------------------------------------------
__global__ void finalize_kernel(const float* __restrict__ partials,
                                float* __restrict__ out) {
    if (threadIdx.x == 0 && blockIdx.x == 0) {
        float s = 0.f;
        for (int i = 0; i < NTILE; ++i) s += partials[i];
        out[0] = -s / (float)M_ROWS;
    }
}

// ---------------------------------------------------------------------------
extern "C" void kernel_launch(void* const* d_in, const int* in_sizes, int n_in,
                              void* d_out, int out_size, void* d_ws, size_t ws_size,
                              hipStream_t stream) {
    const float* feats  = (const float*)d_in[0];     // [8192,128] fp32
    const int*   labels = (const int*)d_in[1];       // [8192] (harness: int)

    // workspace: hi bf16 (2MB) | lo bf16 (2MB) | partials (2KB)
    unsigned short* hi = (unsigned short*)d_ws;
    unsigned short* lo = hi + (size_t)M_ROWS * K_DIM;
    float* partials    = (float*)(lo + (size_t)M_ROWS * K_DIM);

    const int n = M_ROWS * K_DIM;
    split_bf16_kernel<<<(n + 255) / 256, 256, 0, stream>>>(feats, hi, lo, n);
    supcon_loss_kernel<<<NTILE, 256, 0, stream>>>(hi, lo, labels, partials);
    finalize_kernel<<<1, 32, 0, stream>>>(partials, (float*)d_out);
}
